// PointNet_75290776699481
// MI455X (gfx1250) — compile-verified, matrix operand order checked by reading, not run on hardware
//
#include <hip/hip_runtime.h>
#include <hip/hip_bf16.h>
#include <math.h>

#define NPTS 8192
#define KNN  16
#define CW   256
#define HWID 128

typedef __attribute__((ext_vector_type(16))) _Float16 v16h;
typedef __attribute__((ext_vector_type(8)))  _Float16 v8h;
typedef __attribute__((ext_vector_type(8)))  float    v8f;

// Build a 16-half WMMA fragment from two contiguous 16-byte chunks at p and p+16 halves.
__device__ __forceinline__ v16h mk_frag(const _Float16* p) {
  v8h lo = *(const v8h*)p;
  v8h hi = *(const v8h*)(p + 16);
  v16h r;
#pragma unroll
  for (int j = 0; j < 8; ++j) { r[j] = lo[j]; r[j + 8] = hi[j]; }
  return r;
}

// ---------------------------------------------------------------------------
// KNN: one thread per target point; source points tiled through LDS; each
// thread keeps a fully-unrolled sorted top-16 (ascending d2) in registers.
// ---------------------------------------------------------------------------
__global__ __launch_bounds__(256) void knn_kernel(const float* __restrict__ pos,
                                                  int* __restrict__ idx) {
  __shared__ __align__(16) float sp[256 * 3];
  const int i = blockIdx.x * 256 + threadIdx.x;
  const float px = pos[i * 3 + 0];
  const float py = pos[i * 3 + 1];
  const float pz = pos[i * 3 + 2];

  float bd[KNN];
  int   bi[KNN];
#pragma unroll
  for (int j = 0; j < KNN; ++j) { bd[j] = 3.0e38f; bi[j] = 0; }

  for (int base = 0; base < NPTS; base += 256) {
    sp[threadIdx.x]       = pos[base * 3 + threadIdx.x];
    sp[threadIdx.x + 256] = pos[base * 3 + threadIdx.x + 256];
    sp[threadIdx.x + 512] = pos[base * 3 + threadIdx.x + 512];
    __syncthreads();
    for (int j = 0; j < 256; ++j) {
      const float dx = sp[j * 3 + 0] - px;
      const float dy = sp[j * 3 + 1] - py;
      const float dz = sp[j * 3 + 2] - pz;
      const float d2 = dx * dx + dy * dy + dz * dz;
      if (d2 < bd[KNN - 1]) {
        bd[KNN - 1] = d2; bi[KNN - 1] = base + j;
#pragma unroll
        for (int t = KNN - 1; t > 0; --t) {
          if (bd[t] < bd[t - 1]) {
            float td = bd[t]; bd[t] = bd[t - 1]; bd[t - 1] = td;
            int   ti = bi[t]; bi[t] = bi[t - 1]; bi[t - 1] = ti;
          }
        }
      }
    }
    __syncthreads();
  }
#pragma unroll
  for (int j = 0; j < KNN; ++j) idx[i * KNN + j] = bi[j];
}

// ---------------------------------------------------------------------------
// Weight prep: W [Kin][256] f32 -> Wt [256][Kpad] f16 (transposed, zero pad).
// ---------------------------------------------------------------------------
__global__ void convert_wt(const float* __restrict__ W, _Float16* __restrict__ Wt,
                           int Kin, int Kpad, int Cout) {
  int t = blockIdx.x * 256 + threadIdx.x;
  if (t >= Cout * Kpad) return;
  int n = t / Kpad, k = t % Kpad;
  Wt[n * Kpad + k] = (k < Kin) ? (_Float16)W[k * Cout + n] : (_Float16)0.0f;
}

// ---------------------------------------------------------------------------
// Fused PointNetConv layer: one workgroup (8 waves) per point.
//   aP[16][Kpad]  = concat(h[idx], rel) panel (LDS, f16)
//   GEMM1 (WMMA): h1 = relu(aP @ W1t^T + b1) -> LDS f16 [16][256]
//   GEMM2 (WMMA): m  = h1 @ W2t^T            -> per-wave f32 tiles
//   out[i][n] = relu(max_k m[k][n] + b2[n])
// ---------------------------------------------------------------------------
__global__ __launch_bounds__(256) void pointnet_layer(
    const float*    __restrict__ pos,
    const int*      __restrict__ idx,
    const _Float16* __restrict__ hIn,     // null for layer 1 (use pos as features)
    const _Float16* __restrict__ W1t, int Kpad, int cin,
    const float*    __restrict__ b1,
    const _Float16* __restrict__ W2t,
    const float*    __restrict__ b2,
    _Float16*       __restrict__ hOut16,
    float*          __restrict__ hOutF32) {
  __shared__ __align__(16) _Float16 aP[16 * 288];
  __shared__ __align__(16) _Float16 h1[16 * 256];

  const int i      = blockIdx.x;
  const int tid    = threadIdx.x;
  const int lane16 = tid & 15;

  // ---- gather phase ----
  if (cin == 256) {
    const int nb  = tid >> 4;                    // neighbor slot 0..15
    const int src = idx[i * KNN + nb];
    const _Float16* hrow = hIn + src * 256;
    const int c = lane16 * 16;
    *(v8h*)(&aP[nb * Kpad + c])     = *(const v8h*)(hrow + c);
    *(v8h*)(&aP[nb * Kpad + c + 8]) = *(const v8h*)(hrow + c + 8);
    aP[nb * Kpad + 256 + lane16 * 2]     = (_Float16)0.0f;   // zero pad 256..287
    aP[nb * Kpad + 256 + lane16 * 2 + 1] = (_Float16)0.0f;
  } else {
    if (tid < 16) {
      const int nb  = tid;
      const int src = idx[i * KNN + nb];
      const float sx = pos[src * 3 + 0], sy = pos[src * 3 + 1], sz = pos[src * 3 + 2];
      aP[nb * Kpad + 0] = (_Float16)sx;
      aP[nb * Kpad + 1] = (_Float16)sy;
      aP[nb * Kpad + 2] = (_Float16)sz;
      aP[nb * Kpad + 3] = (_Float16)(sx - pos[i * 3 + 0]);
      aP[nb * Kpad + 4] = (_Float16)(sy - pos[i * 3 + 1]);
      aP[nb * Kpad + 5] = (_Float16)(sz - pos[i * 3 + 2]);
#pragma unroll
      for (int c = 6; c < 32; ++c) aP[nb * Kpad + c] = (_Float16)0.0f;
    }
  }
  __syncthreads();
  if (cin == 256 && tid < 16) {                  // rel coords at cols 256..258
    const int nb  = tid;
    const int src = idx[i * KNN + nb];
    aP[nb * Kpad + 256] = (_Float16)(pos[src * 3 + 0] - pos[i * 3 + 0]);
    aP[nb * Kpad + 257] = (_Float16)(pos[src * 3 + 1] - pos[i * 3 + 1]);
    aP[nb * Kpad + 258] = (_Float16)(pos[src * 3 + 2] - pos[i * 3 + 2]);
  }
  __syncthreads();

  const int wave = tid >> 5;                     // 0..7, owns cols [wave*32, wave*32+32)
  const int lane = tid & 31;
  const int lrow = lane & 15;                    // fragment row (A) / col (B,D)
  const int khi  = (lane & 16) ? 8 : 0;          // half-lane K base
  const int n0   = wave * 32;

  // ---- GEMM1: bias-initialized accumulators ----
  v8f acc0, acc1;
  {
    const float bb0 = b1[n0 + lrow];
    const float bb1 = b1[n0 + 16 + lrow];
#pragma unroll
    for (int r = 0; r < 8; ++r) { acc0[r] = bb0; acc1[r] = bb1; }
  }
  const int ksteps1 = Kpad >> 5;
  for (int ks = 0; ks < ksteps1; ++ks) {
    const int kb = ks * 32 + khi;
    v16h A  = mk_frag(&aP[lrow * Kpad + kb]);
    v16h B0 = mk_frag(W1t + (n0 + lrow) * Kpad + kb);
    v16h B1 = mk_frag(W1t + (n0 + 16 + lrow) * Kpad + kb);
    acc0 = __builtin_amdgcn_wmma_f32_16x16x32_f16(false, A, false, B0, (short)0, acc0, false, false);
    acc1 = __builtin_amdgcn_wmma_f32_16x16x32_f16(false, A, false, B1, (short)0, acc1, false, false);
  }
  // ReLU + scatter D tile into h1 LDS panel (f16)
  {
    const int mofs = (lane & 16) ? 8 : 0;
#pragma unroll
    for (int r = 0; r < 8; ++r) {
      const int m = r + mofs;
      h1[m * 256 + n0 + lrow]      = (_Float16)fmaxf(acc0[r], 0.0f);
      h1[m * 256 + n0 + 16 + lrow] = (_Float16)fmaxf(acc1[r], 0.0f);
    }
  }
  __syncthreads();

  // ---- GEMM2 ----
#pragma unroll
  for (int r = 0; r < 8; ++r) { acc0[r] = 0.0f; acc1[r] = 0.0f; }
#pragma unroll
  for (int ks = 0; ks < 8; ++ks) {
    const int kb = ks * 32 + khi;
    v16h A  = mk_frag(&h1[lrow * 256 + kb]);
    v16h B0 = mk_frag(W2t + (n0 + lrow) * 256 + kb);
    v16h B1 = mk_frag(W2t + (n0 + 16 + lrow) * 256 + kb);
    acc0 = __builtin_amdgcn_wmma_f32_16x16x32_f16(false, A, false, B0, (short)0, acc0, false, false);
    acc1 = __builtin_amdgcn_wmma_f32_16x16x32_f16(false, A, false, B1, (short)0, acc1, false, false);
  }

  // ---- segment max over the 16 neighbor rows ----
  float m0 = -3.0e38f, m1 = -3.0e38f;
#pragma unroll
  for (int r = 0; r < 8; ++r) { m0 = fmaxf(m0, acc0[r]); m1 = fmaxf(m1, acc1[r]); }
  m0 = fmaxf(m0, __shfl_xor(m0, 16, 32));        // combine M=0..7 with M=8..15
  m1 = fmaxf(m1, __shfl_xor(m1, 16, 32));
  if (lane < 16) {
    const int c0 = n0 + lrow;
    const int c1 = n0 + 16 + lrow;
    const float o0 = fmaxf(m0 + b2[c0], 0.0f);
    const float o1 = fmaxf(m1 + b2[c1], 0.0f);
    hOut16[i * 256 + c0] = (_Float16)o0;
    hOut16[i * 256 + c1] = (_Float16)o1;
    if (hOutF32) { hOutF32[i * 256 + c0] = o0; hOutF32[i * 256 + c1] = o1; }
  }
}

// ---------------------------------------------------------------------------
// Classifier head: one block (128 threads) per point.
// ---------------------------------------------------------------------------
__global__ __launch_bounds__(128) void classifier_kernel(
    const float* __restrict__ hf,
    const float* __restrict__ W1, const float* __restrict__ b1,
    const float* __restrict__ W2, const float* __restrict__ b2,
    const float* __restrict__ W3, const float* __restrict__ b3,
    float* __restrict__ out) {
  __shared__ float sh[256];
  __shared__ float y1[128];
  __shared__ float y2[128];
  __shared__ float red[128];
  const int i = blockIdx.x, t = threadIdx.x;
  sh[t]       = hf[i * 256 + t];
  sh[t + 128] = hf[i * 256 + 128 + t];
  __syncthreads();
  float a = b1[t];
  for (int c = 0; c < 256; ++c) a += sh[c] * W1[c * 128 + t];
  y1[t] = fmaxf(a, 0.0f);
  __syncthreads();
  float a2 = b2[t];
  for (int c = 0; c < 128; ++c) a2 += y1[c] * W2[c * 128 + t];
  y2[t] = fmaxf(a2, 0.0f);
  __syncthreads();
  red[t] = y2[t] * W3[t];
  __syncthreads();
  for (int s = 64; s > 0; s >>= 1) {
    if (t < s) red[t] += red[t + s];
    __syncthreads();
  }
  if (t == 0) out[i] = 1.0f / (1.0f + expf(-(red[0] + b3[0])));
}

// ---------------------------------------------------------------------------
// Workspace layout (bytes, all 256-aligned)
// ---------------------------------------------------------------------------
#define OFF_IDX   ((size_t)0)                                    // 8192*16*4
#define OFF_H16A  (OFF_IDX  + (size_t)NPTS * KNN * 4)            // 4 MB
#define OFF_H16B  (OFF_H16A + (size_t)NPTS * 256 * 2)            // 4 MB
#define OFF_HF32  (OFF_H16B + (size_t)NPTS * 256 * 2)            // 8 MB
#define OFF_W1T1  (OFF_HF32 + (size_t)NPTS * 256 * 4)            // 256*32*2
#define OFF_W1T2  (OFF_W1T1 + (size_t)256 * 32 * 2)              // 256*288*2
#define OFF_W1T3  (OFF_W1T2 + (size_t)256 * 288 * 2)
#define OFF_W2T1  (OFF_W1T3 + (size_t)256 * 288 * 2)             // 256*256*2 x3
#define OFF_W2T2  (OFF_W2T1 + (size_t)256 * 256 * 2)
#define OFF_W2T3  (OFF_W2T2 + (size_t)256 * 256 * 2)

extern "C" void kernel_launch(void* const* d_in, const int* in_sizes, int n_in,
                              void* d_out, int out_size, void* d_ws, size_t ws_size,
                              hipStream_t stream) {
  (void)in_sizes; (void)n_in; (void)out_size; (void)ws_size;
  const float* pos  = (const float*)d_in[0];
  const float* c1W1 = (const float*)d_in[1];
  const float* c1b1 = (const float*)d_in[2];
  const float* c1W2 = (const float*)d_in[3];
  const float* c1b2 = (const float*)d_in[4];
  const float* c2W1 = (const float*)d_in[5];
  const float* c2b1 = (const float*)d_in[6];
  const float* c2W2 = (const float*)d_in[7];
  const float* c2b2 = (const float*)d_in[8];
  const float* c3W1 = (const float*)d_in[9];
  const float* c3b1 = (const float*)d_in[10];
  const float* c3W2 = (const float*)d_in[11];
  const float* c3b2 = (const float*)d_in[12];
  const float* hW1  = (const float*)d_in[13];
  const float* hb1  = (const float*)d_in[14];
  const float* hW2  = (const float*)d_in[15];
  const float* hb2  = (const float*)d_in[16];
  const float* hW3  = (const float*)d_in[17];
  const float* hb3  = (const float*)d_in[18];

  char* ws = (char*)d_ws;
  int*      idx  = (int*)     (ws + OFF_IDX);
  _Float16* h16a = (_Float16*)(ws + OFF_H16A);
  _Float16* h16b = (_Float16*)(ws + OFF_H16B);
  float*    hf32 = (float*)   (ws + OFF_HF32);
  _Float16* w1t1 = (_Float16*)(ws + OFF_W1T1);
  _Float16* w1t2 = (_Float16*)(ws + OFF_W1T2);
  _Float16* w1t3 = (_Float16*)(ws + OFF_W1T3);
  _Float16* w2t1 = (_Float16*)(ws + OFF_W2T1);
  _Float16* w2t2 = (_Float16*)(ws + OFF_W2T2);
  _Float16* w2t3 = (_Float16*)(ws + OFF_W2T3);

  // KNN graph
  knn_kernel<<<NPTS / 256, 256, 0, stream>>>(pos, idx);

  // Weight prep (transpose + f16 + zero pad)
  convert_wt<<<(256 * 32 + 255) / 256, 256, 0, stream>>>(c1W1, w1t1, 6, 32, 256);
  convert_wt<<<(256 * 288 + 255) / 256, 256, 0, stream>>>(c2W1, w1t2, 259, 288, 256);
  convert_wt<<<(256 * 288 + 255) / 256, 256, 0, stream>>>(c3W1, w1t3, 259, 288, 256);
  convert_wt<<<(256 * 256 + 255) / 256, 256, 0, stream>>>(c1W2, w2t1, 256, 256, 256);
  convert_wt<<<(256 * 256 + 255) / 256, 256, 0, stream>>>(c2W2, w2t2, 256, 256, 256);
  convert_wt<<<(256 * 256 + 255) / 256, 256, 0, stream>>>(c3W2, w2t3, 256, 256, 256);

  // Fused conv layers (one workgroup per point, WMMA GEMMs)
  pointnet_layer<<<NPTS, 256, 0, stream>>>(pos, idx, nullptr, w1t1, 32, 3,
                                           c1b1, w2t1, c1b2, h16a, nullptr);
  pointnet_layer<<<NPTS, 256, 0, stream>>>(pos, idx, h16a, w1t2, 288, 256,
                                           c2b1, w2t2, c2b2, h16b, nullptr);
  pointnet_layer<<<NPTS, 256, 0, stream>>>(pos, idx, h16b, w1t3, 288, 256,
                                           c3b1, w2t3, c3b2, h16a, hf32);

  // MLP head + sigmoid
  classifier_kernel<<<NPTS, 128, 0, stream>>>(hf32, hW1, hb1, hW2, hb2, hW3, hb3,
                                              (float*)d_out);
}
